// CrossAttention_32066225832755
// MI455X (gfx1250) — matrix operand order repaired and checked by reading, not compiled, not run on hardware
//
#include <hip/hip_runtime.h>
#include <hip/hip_bf16.h>

typedef __attribute__((ext_vector_type(16))) _Float16 v16h;
typedef __attribute__((ext_vector_type(8)))  _Float16 v8h;
typedef __attribute__((ext_vector_type(8)))  float    v8f;

constexpr int C_D    = 768;
constexpr int C_H    = 6;
constexpr int C_DH   = 128;
constexpr int C_NQ   = 512;
constexpr int C_G    = 64;
constexpr int C_R    = 64;
constexpr int C_MID  = 4096;
constexpr int C_VREP = 8;

#define WMMA_F16(a, b, c) \
  __builtin_amdgcn_wmma_f32_16x16x32_f16(false, (a), false, (b), (short)0, (c), false, false)

// ---------------------------------------------------------------------------
// Fragment loaders (CDNA5 16x16x32 f16 layouts, CDNA5 ISA 7.12.2)
// A (16xK tile @ k_base, row-major src): lane L -> row (L&15),
//   k = (e&7) + 16*(e>>3) + 8*(L>>4)  => two contiguous 8-half (16B) runs.
// ---------------------------------------------------------------------------
static __device__ __forceinline__ v16h wmma_af(const _Float16* p, int ld,
                                               int row_base, int k_base) {
  int lane = threadIdx.x & 31;
  const _Float16* r =
      p + (size_t)(row_base + (lane & 15)) * ld + k_base + ((lane >> 4) << 3);
  v8h lo = *(const v8h*)(r);
  v8h hi = *(const v8h*)(r + 16);
  v16h a;
#pragma unroll
  for (int i = 0; i < 8; ++i) { a[i] = lo[i]; a[i + 8] = hi[i]; }
  return a;
}

// B (32x16 tile, row-major src with leading dim ld): lane L -> column (L&15),
//   k = (e&7) + 16*(e>>3) + 8*(L>>4)  (same k order as A).
static __device__ __forceinline__ v16h wmma_bf(const _Float16* p, int ld,
                                               int k_base, int col_base) {
  int lane = threadIdx.x & 31;
  const _Float16* q = p + (size_t)(k_base + ((lane >> 4) << 3)) * ld + col_base + (lane & 15);
  v16h b;
#pragma unroll
  for (int e = 0; e < 8; ++e) { b[e] = q[(size_t)e * ld]; b[e + 8] = q[(size_t)(e + 16) * ld]; }
  return b;
}

// ---------------------------------------------------------------------------
// fp32 -> f16 conversion
// ---------------------------------------------------------------------------
__global__ void cvt_kernel(const float* __restrict__ in, _Float16* __restrict__ out,
                           size_t n) {
  size_t i = (size_t)blockIdx.x * blockDim.x + threadIdx.x;
  if (i < n) out[i] = (_Float16)in[i];
}

// Wc (G, D, R) fp32 -> WcT (G, R, D) f16
__global__ void cvt_wct_kernel(const float* __restrict__ in, _Float16* __restrict__ out) {
  size_t i = (size_t)blockIdx.x * blockDim.x + threadIdx.x;
  size_t total = (size_t)C_G * C_D * C_R;
  if (i >= total) return;
  int r = (int)(i % C_R);
  size_t t = i / C_R;
  int d = (int)(t % C_D);
  int g = (int)(t / C_D);
  out[((size_t)g * C_R + r) * C_D + d] = (_Float16)in[i];
}

// ---------------------------------------------------------------------------
// GEMM: C(f32, MxN) = A(f16, MxK, rm) * B(f16, KxN, rm)
// block = 128 threads (4 waves). Wave computes 32x64 (2 M-tiles x 4 N-tiles).
// grid = (M/32, N/256)
// ---------------------------------------------------------------------------
__global__ void gemm_kernel(const _Float16* __restrict__ A,
                            const _Float16* __restrict__ B,
                            float* __restrict__ C, int M, int K, int N) {
  int wave = threadIdx.x >> 5;
  int lane = threadIdx.x & 31;
  int row0 = blockIdx.x * 32;
  int col0 = blockIdx.y * 256 + wave * 64;
  v8f acc[2][4] = {};
  for (int k = 0; k < K; k += 32) {
    v16h a0 = wmma_af(A, K, row0, k);
    v16h a1 = wmma_af(A, K, row0 + 16, k);
#pragma unroll
    for (int n = 0; n < 4; ++n) {
      v16h b = wmma_bf(B, N, k, col0 + n * 16);
      acc[0][n] = WMMA_F16(a0, b, acc[0][n]);
      acc[1][n] = WMMA_F16(a1, b, acc[1][n]);
    }
  }
  int ccol = lane & 15;
  int r0 = (lane >> 4) << 3;
#pragma unroll
  for (int m = 0; m < 2; ++m)
#pragma unroll
    for (int n = 0; n < 4; ++n)
#pragma unroll
      for (int i = 0; i < 8; ++i)
        C[(size_t)(row0 + m * 16 + r0 + i) * N + col0 + n * 16 + ccol] = acc[m][n][i];
}

// ---------------------------------------------------------------------------
// Row LayerNorm: in f32 (rows x C) -> out f16.  block=256, grid=rows
// ---------------------------------------------------------------------------
__global__ void ln_kernel(const float* __restrict__ in, const float* __restrict__ gamma,
                          const float* __restrict__ beta, _Float16* __restrict__ out,
                          int C) {
  int t = threadIdx.x;
  size_t base = (size_t)blockIdx.x * C;
  float s = 0.f, s2 = 0.f;
  for (int j = t; j < C; j += 256) {
    float v = in[base + j];
    s += v; s2 += v * v;
  }
  __shared__ float rs[256], rq[256];
  rs[t] = s; rq[t] = s2;
  __syncthreads();
  for (int off = 128; off > 0; off >>= 1) {
    if (t < off) { rs[t] += rs[t + off]; rq[t] += rq[t + off]; }
    __syncthreads();
  }
  float mu = rs[0] / C;
  float var = rq[0] / C - mu * mu;
  float rstd = rsqrtf(var + 1e-5f);
  for (int j = t; j < C; j += 256)
    out[base + j] = (_Float16)((in[base + j] - mu) * rstd * gamma[j] + beta[j]);
}

// ---------------------------------------------------------------------------
// Attention scores + softmax.
// grid = (NQ/16, H, B), block = 64 (2 waves).
// sim[16x256] = q_tile (16x128) * k^T, chunked over DH in 32s via LDS staging.
// attn written as f16 (B,H,NQ,N).
// ---------------------------------------------------------------------------
__global__ void attn_kernel(const _Float16* __restrict__ q16,
                            const _Float16* __restrict__ k16,
                            _Float16* __restrict__ attn, int Nk) {
  const int qt = blockIdx.x, h = blockIdx.y, b = blockIdx.z;
  const int t = threadIdx.x, wave = t >> 5, lane = t & 31;
  __shared__ _Float16 kT[32][264];      // one 32-wide d-chunk of k^T
  __shared__ float simbuf[16][256];
  v8f acc[8] = {};
  const _Float16* qbase = q16 + (size_t)b * C_NQ * C_D;
  const _Float16* kbase = k16 + (size_t)b * Nk * C_D;

  for (int kc = 0; kc < 4; ++kc) {
    __syncthreads();
    // stage kT chunk: kT[d_local][j] = k[b, j, h*128 + kc*32 + d_local]
    for (int it = 0; it < 16; ++it) {
      int c = t + it * 64;              // [0,1024) v8h chunks
      int j = c >> 2;
      int d0 = (c & 3) * 8;
      v8h v = *(const v8h*)(kbase + (size_t)j * C_D + h * C_DH + kc * 32 + d0);
#pragma unroll
      for (int e = 0; e < 8; ++e) kT[d0 + e][j] = v[e];
    }
    __syncthreads();
    v16h a = wmma_af(qbase, C_D, qt * 16, h * C_DH + kc * 32);
#pragma unroll
    for (int i = 0; i < 8; ++i) {
      int jt = wave * 8 + i;
      v16h bfr;
#pragma unroll
      for (int e = 0; e < 8; ++e) {
        bfr[e]     = kT[((lane >> 4) << 3) + e][jt * 16 + (lane & 15)];
        bfr[e + 8] = kT[((lane >> 4) << 3) + 16 + e][jt * 16 + (lane & 15)];
      }
      acc[i] = WMMA_F16(a, bfr, acc[i]);
    }
  }
  // scale = DH^-0.25 applied to q and k => product DH^-0.5 = 1/sqrt(128)
  const float scale = 0.08838834764831845f;
  {
    int ccol = lane & 15, r0 = (lane >> 4) << 3;
#pragma unroll
    for (int i = 0; i < 8; ++i) {
      int jt = wave * 8 + i;
#pragma unroll
      for (int m = 0; m < 8; ++m)
        simbuf[r0 + m][jt * 16 + ccol] = acc[i][m] * scale;
    }
  }
  __syncthreads();
  if (t < 16) {
    float mx = -1e30f;
    for (int j = 0; j < Nk; ++j) mx = fmaxf(mx, simbuf[t][j]);
    float sum = 0.f;
    for (int j = 0; j < Nk; ++j) {
      float e = __expf(simbuf[t][j] - mx);
      simbuf[t][j] = e; sum += e;
    }
    float inv = 1.f / sum;
    _Float16* arow = attn + (size_t)(((b * C_H + h) * C_NQ) + qt * 16 + t) * Nk;
    for (int j = 0; j < Nk; ++j) arow[j] = (_Float16)(simbuf[t][j] * inv);
  }
}

// ---------------------------------------------------------------------------
// Fused grouped conv + LN(v) + attention-weighted sum.
// grid = (G, B), block = 256 (8 waves).
// Per 32-row j-tile: v = v1g @ Wc[g]^T (WMMA) -> LDS f16; LN stats; then
//   out_d = gamma_d * (sum_j a_j*rstd_j*v_jd) - gamma_d * (sum_j a_j*mu_j*rstd_j)
//           + beta_d            (uses softmax sum_j a_j == 1)
// second reduction also done with WMMA (A' = attn*rstd, M=8 padded to 16).
// ---------------------------------------------------------------------------
__global__ void vout_kernel(const _Float16* __restrict__ v116,
                            const _Float16* __restrict__ wct16,
                            const _Float16* __restrict__ attn,
                            const float* __restrict__ gv2,
                            const float* __restrict__ bv2,
                            _Float16* __restrict__ o16, int Nk) {
  const int g = blockIdx.x, b = blockIdx.y;
  const int t = threadIdx.x, wave = t >> 5, lane = t & 31;
  __shared__ _Float16 vt16[32][776];    // raw v tile (32 j-rows x 768)
  __shared__ float rstd_s[32], murstd_s[32];
  __shared__ float csum[C_H][C_VREP];
  if (t < C_H * C_VREP) csum[t / C_VREP][t % C_VREP] = 0.f;

  v8f out_acc[6] = {};
  const _Float16* v1b = v116 + (size_t)b * Nk * C_MID;
  const _Float16* wcg = wct16 + (size_t)g * C_R * C_D;
  const int ntiles = Nk / 32;

  for (int nt = 0; nt < ntiles; ++nt) {
    __syncthreads();
    // --- phase 1: grouped conv tile (M=32, K=64, N=768) ---
    v16h a[2][2];
#pragma unroll
    for (int mt = 0; mt < 2; ++mt)
#pragma unroll
      for (int kc = 0; kc < 2; ++kc)
        a[mt][kc] = wmma_af(v1b, C_MID, nt * 32 + mt * 16, g * C_R + kc * 32);
#pragma unroll
    for (int i = 0; i < 12; ++i) {
      int u = wave * 12 + i;            // [0,96): 2 M-tiles x 48 N-tiles
      int mt = u / 48;
      int dt = u % 48;
      v8f acc = {};
#pragma unroll
      for (int kc = 0; kc < 2; ++kc) {
        v16h bfr = wmma_bf(wcg, C_D, kc * 32, dt * 16);
        acc = WMMA_F16(a[mt][kc], bfr, acc);
      }
      int ccol = dt * 16 + (lane & 15);
      int r0 = mt * 16 + ((lane >> 4) << 3);
#pragma unroll
      for (int m = 0; m < 8; ++m) vt16[r0 + m][ccol] = (_Float16)acc[m];
    }
    __syncthreads();
    // --- phase 2: LN stats per j-row ---
    if (t < 32) {
      float s = 0.f, s2 = 0.f;
      for (int c = 0; c < C_D; ++c) {
        float v = (float)vt16[t][c];
        s += v; s2 += v * v;
      }
      float mu = s / C_D;
      float var = s2 / C_D - mu * mu;
      float rs = rsqrtf(var + 1e-5f);
      rstd_s[t] = rs;
      murstd_s[t] = mu * rs;
    }
    __syncthreads();
    // mean-correction partial sums: c_{h,r} += sum_j a_j * mu_j * rstd_j
    if (t < C_H * C_VREP) {
      int h = t / C_VREP, r = t % C_VREP;
      const _Float16* arow =
          attn + (size_t)(((b * C_H + h) * C_NQ) + r * C_G + g) * Nk + nt * 32;
      float p = 0.f;
      for (int j = 0; j < 32; ++j) p += (float)arow[j] * murstd_s[j];
      csum[h][r] += p;
    }
    // --- phase 3: weighted sum via WMMA, accumulate over j-tiles ---
#pragma unroll
    for (int i = 0; i < 6; ++i) {
      int u = wave * 6 + i;             // [0,48): (h, dt)
      int h = u >> 3, dt = u & 7;
      v16h ap = {};
      int m = lane & 15;
      int k0 = (lane >> 4) << 3;
      if (m < C_VREP) {
        const _Float16* arow =
            attn + (size_t)(((b * C_H + h) * C_NQ) + m * C_G + g) * Nk + nt * 32;
        v8h lo = *(const v8h*)(arow + k0);
        v8h hi = *(const v8h*)(arow + k0 + 16);
#pragma unroll
        for (int e = 0; e < 8; ++e) {
          ap[e]     = (_Float16)((float)lo[e] * rstd_s[k0 + e]);
          ap[e + 8] = (_Float16)((float)hi[e] * rstd_s[k0 + 16 + e]);
        }
      }
      v16h bfr;
#pragma unroll
      for (int e = 0; e < 8; ++e) {
        bfr[e]     = vt16[k0 + e][h * C_DH + dt * 16 + m];
        bfr[e + 8] = vt16[k0 + 16 + e][h * C_DH + dt * 16 + m];
      }
      out_acc[i] = WMMA_F16(ap, bfr, out_acc[i]);
    }
  }
  __syncthreads();
  // --- epilogue: apply gamma/beta + mean correction, write f16 ---
  if (lane < 16) {
#pragma unroll
    for (int i = 0; i < 6; ++i) {
      int u = wave * 6 + i;
      int h = u >> 3, dt = u & 7;
      int c = h * C_DH + dt * 16 + lane;
      float gm = gv2[c], bt = bv2[c];
#pragma unroll
      for (int r = 0; r < 8; ++r) {
        float val = gm * (out_acc[i][r] - csum[h][r]) + bt;
        o16[((size_t)b * C_NQ + r * C_G + g) * C_D + c] = (_Float16)val;
      }
    }
  }
}

// ---------------------------------------------------------------------------
extern "C" void kernel_launch(void* const* d_in, const int* in_sizes, int n_in,
                              void* d_out, int out_size, void* d_ws, size_t ws_size,
                              hipStream_t stream) {
  const float* x    = (const float*)d_in[0];
  const float* ctx  = (const float*)d_in[1];
  const float* Wq   = (const float*)d_in[2];
  const float* gq   = (const float*)d_in[3];
  const float* bq   = (const float*)d_in[4];
  const float* Wk   = (const float*)d_in[5];
  const float* gk   = (const float*)d_in[6];
  const float* bk   = (const float*)d_in[7];
  const float* Wv1  = (const float*)d_in[8];
  const float* gv1  = (const float*)d_in[9];
  const float* bv1  = (const float*)d_in[10];
  const float* Wc   = (const float*)d_in[11];
  const float* gv2  = (const float*)d_in[12];
  const float* bv2  = (const float*)d_in[13];
  const float* Wout = (const float*)d_in[14];

  int Bv = in_sizes[0] / (C_NQ * C_D);   // 4
  int Nk = in_sizes[1] / (Bv * C_D);     // 256

  char* p = (char*)d_ws;
  auto carve = [&](size_t elems, size_t esz) -> void* {
    void* r = (void*)p;
    p += (elems * esz + 255) & ~(size_t)255;
    return r;
  };
  _Float16* x16    = (_Float16*)carve((size_t)Bv * C_NQ * C_D, 2);
  _Float16* c16    = (_Float16*)carve((size_t)Bv * Nk * C_D, 2);
  _Float16* wq16   = (_Float16*)carve((size_t)C_D * C_D, 2);
  _Float16* wk16   = (_Float16*)carve((size_t)C_D * C_D, 2);
  _Float16* wv116  = (_Float16*)carve((size_t)C_D * C_MID, 2);
  _Float16* wct16  = (_Float16*)carve((size_t)C_G * C_R * C_D, 2);
  _Float16* wout16 = (_Float16*)carve((size_t)C_D * C_D, 2);
  float*    qpre   = (float*)carve((size_t)Bv * C_NQ * C_D, 4);
  float*    kpre   = (float*)carve((size_t)Bv * Nk * C_D, 4);
  float*    v1pre  = (float*)carve((size_t)Bv * Nk * C_MID, 4);
  _Float16* q16    = (_Float16*)carve((size_t)Bv * C_NQ * C_D, 2);
  _Float16* k16    = (_Float16*)carve((size_t)Bv * Nk * C_D, 2);
  _Float16* v116   = (_Float16*)carve((size_t)Bv * Nk * C_MID, 2);
  _Float16* attn16 = (_Float16*)carve((size_t)Bv * C_H * C_NQ * Nk, 2);
  _Float16* o16    = (_Float16*)carve((size_t)Bv * C_NQ * C_D, 2);

  auto cvt = [&](const float* in, _Float16* out, size_t n) {
    cvt_kernel<<<dim3((unsigned)((n + 255) / 256)), dim3(256), 0, stream>>>(in, out, n);
  };
  cvt(x, x16, (size_t)Bv * C_NQ * C_D);
  cvt(ctx, c16, (size_t)Bv * Nk * C_D);
  cvt(Wq, wq16, (size_t)C_D * C_D);
  cvt(Wk, wk16, (size_t)C_D * C_D);
  cvt(Wv1, wv116, (size_t)C_D * C_MID);
  cvt(Wout, wout16, (size_t)C_D * C_D);
  {
    size_t n = (size_t)C_G * C_D * C_R;
    cvt_wct_kernel<<<dim3((unsigned)((n + 255) / 256)), dim3(256), 0, stream>>>(Wc, wct16);
  }

  // q = LN(x @ Wq)
  gemm_kernel<<<dim3(Bv * C_NQ / 32, C_D / 256), 128, 0, stream>>>(
      x16, wq16, qpre, Bv * C_NQ, C_D, C_D);
  ln_kernel<<<Bv * C_NQ, 256, 0, stream>>>(qpre, gq, bq, q16, C_D);
  // k = LN(context @ Wk)
  gemm_kernel<<<dim3(Bv * Nk / 32, C_D / 256), 128, 0, stream>>>(
      c16, wk16, kpre, Bv * Nk, C_D, C_D);
  ln_kernel<<<Bv * Nk, 256, 0, stream>>>(kpre, gk, bk, k16, C_D);
  // v1 = LN(context @ Wv1)
  gemm_kernel<<<dim3(Bv * Nk / 32, C_MID / 256), 128, 0, stream>>>(
      c16, wv116, v1pre, Bv * Nk, C_D, C_MID);
  ln_kernel<<<Bv * Nk, 256, 0, stream>>>(v1pre, gv1, bv1, v116, C_MID);
  // attention scores + softmax
  attn_kernel<<<dim3(C_NQ / 16, C_H, Bv), 64, 0, stream>>>(q16, k16, attn16, Nk);
  // fused grouped conv + LN(v) + attention-weighted sum
  vout_kernel<<<dim3(C_G, Bv), 256, 0, stream>>>(v116, wct16, attn16, gv2, bv2, o16, Nk);
  // final projection -> d_out (fp32)
  gemm_kernel<<<dim3(Bv * C_NQ / 32, C_D / 256), 128, 0, stream>>>(
      o16, wout16, (float*)d_out, Bv * C_NQ, C_D, C_D);

  (void)n_in; (void)out_size; (void)ws_size;
}
